// QuantLinear_83064667505244
// MI455X (gfx1250) — matrix operand order repaired and checked by reading, not compiled, not run on hardware
//
#include <hip/hip_runtime.h>

typedef __attribute__((ext_vector_type(16))) __bf16        v16bf;
typedef __attribute__((ext_vector_type(8)))  float         v8f;
typedef __attribute__((ext_vector_type(4))) unsigned int   v4u;
typedef __attribute__((ext_vector_type(2))) unsigned int   v2u;

#define BM  128
#define BN  256
#define KC  64
#define XLD 72   // padded LDS row stride (bf16): 144B = 36 dword-banks -> 16-row reads hit all 64 banks
#define WLD 72

// two f32 -> packed bf16x2, round-half-up, one v_perm_b32 for the pack
__device__ __forceinline__ unsigned pk_bf16(float lo, float hi) {
    const unsigned ulo = __float_as_uint(lo) + 0x8000u;
    const unsigned uhi = __float_as_uint(hi) + 0x8000u;
    // dst bytes: {ulo[2],ulo[3],uhi[2],uhi[3]} -> [lo_bf16 | hi_bf16<<16]
    return __builtin_amdgcn_perm(uhi, ulo, 0x07060302u);
}

union Frag { v16bf bf; v4u q[2]; };

__global__ __launch_bounds__(256)
void qlinear3_wmma_bf16(const float* __restrict__ x,
                        const int*   __restrict__ qw,
                        const float* __restrict__ scales,
                        const float* __restrict__ zeros,
                        const float* __restrict__ bias,
                        float*       __restrict__ out,
                        int M, int K, int N) {
    __shared__ __align__(16) unsigned short lds_x[2][BM * XLD]; //  36 KB
    __shared__ __align__(16) unsigned short lds_w[2][BN * WLD]; //  72 KB

    const int t     = threadIdx.x;
    const int lane  = t & 31;
    const int wave  = t >> 5;
    const int waveM = wave >> 2;   // 0..1 : 64 rows of M each
    const int waveN = wave & 3;    // 0..3 : 64 cols of N each
    const int m0    = blockIdx.y * BM;
    const int n0    = blockIdx.x * BN;

    const int qrow_stride = (K >> 5) * 3;

    // every thread owns one of the 256 weight rows of this block
    const float wscale = scales[n0 + t];
    const float wzero  = zeros[n0 + t];
    const int*  qrow0  = qw + (size_t)(n0 + t) * qrow_stride;

    const int cg = t & 15;   // float4 column in 64-wide k chunk
    const int rg = t >> 4;   // base x row 0..15

    v8f acc[4][4];
#pragma unroll
    for (int i = 0; i < 4; ++i)
#pragma unroll
        for (int j = 0; j < 4; ++j)
#pragma unroll
            for (int e = 0; e < 8; ++e) acc[i][j][e] = 0.0f;

    const int nk = K / KC;

    // ---------------- staging (one K chunk -> LDS buffer p) ----------------
    auto stage = [&](int p, int k0) {
        // X tile: BM x KC f32 -> bf16 (coalesced 256B row segments)
        const float* xbase = x + (size_t)m0 * K + k0 + cg * 4;
#pragma unroll
        for (int i = 0; i < 8; ++i) {
            const int row = rg + i * 16;
            const float4 xv = *(const float4*)(xbase + (size_t)row * K);
            v2u pk;
            pk.x = pk_bf16(xv.x, xv.y);
            pk.y = pk_bf16(xv.z, xv.w);
            *(v2u*)(&lds_x[p][row * XLD + cg * 4]) = pk;
        }
        // W tile: 256 rows x KC, unpack 3-bit -> dequant -> bf16
        const int* qrow = qrow0 + (k0 >> 5) * 3;
        __builtin_prefetch(qrow + 6, 0, 0);   // next chunk of the packed stream
#pragma unroll
        for (int g = 0; g < 2; ++g) {
            const unsigned w0 = (unsigned)qrow[g * 3 + 0];
            const unsigned w1 = (unsigned)qrow[g * 3 + 1];
            const unsigned w2 = (unsigned)qrow[g * 3 + 2];
            int v[32];
#pragma unroll
            for (int j = 0; j < 10; ++j) v[j]      = (int)((w0 >> (3 * j)) & 7u);
            v[10] = (int)(((w0 >> 30) & 3u) | ((w1 & 1u) << 2));
#pragma unroll
            for (int j = 0; j < 10; ++j) v[11 + j] = (int)((w1 >> (3 * j + 1)) & 7u);
            v[21] = (int)(((w1 >> 31) & 1u) | ((w2 & 3u) << 1));
#pragma unroll
            for (int j = 0; j < 10; ++j) v[22 + j] = (int)((w2 >> (3 * j + 2)) & 7u);

            unsigned short* dst = &lds_w[p][t * WLD + g * 32];
#pragma unroll
            for (int j = 0; j < 4; ++j) {
                v4u qv;
#pragma unroll
                for (int e = 0; e < 4; ++e) {
                    const int   i0 = 8 * j + 2 * e;
                    const float flo = (float)v[i0]     * wscale - wzero;
                    const float fhi = (float)v[i0 + 1] * wscale - wzero;
                    qv[e] = pk_bf16(flo, fhi);
                }
                *(v4u*)(dst + j * 8) = qv;
            }
        }
    };

    // ---------------- compute (one K chunk from LDS buffer p) ----------------
    const int half = lane >> 4;
    const int l16  = lane & 15;
    auto compute = [&](int p) {
#pragma unroll
        for (int kk = 0; kk < 2; ++kk) {
            Frag a[4], b[4];
#pragma unroll
            for (int mt = 0; mt < 4; ++mt) {
                // A 16x32: lanes 0-15 K0..15; lanes 16-31 K16..31 (contiguous)
                const int row = waveM * 64 + mt * 16 + l16;
                const unsigned short* src = &lds_x[p][row * XLD + kk * 32 + half * 16];
                a[mt].q[0] = *(const v4u*)(src);
                a[mt].q[1] = *(const v4u*)(src + 8);
            }
#pragma unroll
            for (int nt = 0; nt < 4; ++nt) {
                // B 32x16: lanes 0-15 K0..15; lanes 16-31 K16..31 (contiguous)
                const int row = waveN * 64 + nt * 16 + l16;
                const unsigned short* src = &lds_w[p][row * WLD + kk * 32 + half * 16];
                b[nt].q[0] = *(const v4u*)(src);
                b[nt].q[1] = *(const v4u*)(src + 8);
            }
#pragma unroll
            for (int mt = 0; mt < 4; ++mt)
#pragma unroll
                for (int nt = 0; nt < 4; ++nt)
                    acc[mt][nt] = __builtin_amdgcn_wmma_f32_16x16x32_bf16(
                        false, a[mt].bf, false, b[nt].bf,
                        (short)0, acc[mt][nt], false, false);
        }
    };

    // ---------------- pipelined main loop: 1 barrier / chunk ----------------
    stage(0, 0);
    for (int i = 0; i < nk; ++i) {
        __syncthreads();
        if (i + 1 < nk) stage((i + 1) & 1, (i + 1) * KC);
        compute(i & 1);
    }

    // ---------------- epilogue: y = acc + bias ----------------
#pragma unroll
    for (int nt = 0; nt < 4; ++nt) {
        const int n  = n0 + waveN * 64 + nt * 16 + l16;
        const float bn = bias[n];
#pragma unroll
        for (int mt = 0; mt < 4; ++mt) {
            const int mbase = m0 + waveM * 64 + mt * 16 + half * 8;
#pragma unroll
            for (int r = 0; r < 8; ++r) {
                out[(size_t)(mbase + r) * N + n] = acc[mt][nt][r] + bn;
            }
        }
    }
}

extern "C" void kernel_launch(void* const* d_in, const int* in_sizes, int n_in,
                              void* d_out, int out_size, void* d_ws, size_t ws_size,
                              hipStream_t stream) {
    const float* x      = (const float*)d_in[0];
    const int*   qw     = (const int*)d_in[1];
    const float* scales = (const float*)d_in[2];
    const float* zeros  = (const float*)d_in[3];
    const float* bias   = (const float*)d_in[4];
    float*       out    = (float*)d_out;

    const int N = in_sizes[4];
    const int K = (int)(((long)in_sizes[1] / N) * 32 / 3);
    const int M = (int)((long)in_sizes[0] / K);

    dim3 grid(N / BN, M / BM);
    qlinear3_wmma_bf16<<<grid, 256, 0, stream>>>(x, qw, scales, zeros, bias, out, M, K, N);
}
